// GATModel_7378753815012
// MI455X (gfx1250) — matrix operand order repaired and checked by reading, not compiled, not run on hardware
//
#include <hip/hip_runtime.h>
#include <hip/hip_bf16.h>
#include <math.h>
#include <stdint.h>

typedef __attribute__((ext_vector_type(16))) __bf16 v16bf;
typedef __attribute__((ext_vector_type(8)))  float  v8f;

#define HEADS 8

// ---------------- helpers ----------------

__device__ __forceinline__ unsigned short f2bf(float f) {
  unsigned int u = __float_as_uint(f);
  u += 0x7FFFu + ((u >> 16) & 1u);          // round-to-nearest-even
  return (unsigned short)(u >> 16);
}

__device__ __forceinline__ float eluf(float v) {
  return v > 0.0f ? v : (expf(v) - 1.0f);
}

__device__ __forceinline__ void atomicMaxF32(float* addr, float val) {
  // signed-max / unsigned-min trick; works with -inf init and mixed signs
  if (!signbit(val)) atomicMax((int*)addr, __float_as_int(val));
  else               atomicMin((unsigned int*)addr, __float_as_uint(val));
}

// ---------------- utility kernels ----------------

__global__ void fill_f32(float* __restrict__ p, float v, size_t n) {
  size_t i = (size_t)blockIdx.x * blockDim.x + threadIdx.x;
  size_t s = (size_t)gridDim.x * blockDim.x;
  for (; i < n; i += s) p[i] = v;
}

__global__ void pack_bf16(const float* __restrict__ a,
                          unsigned short* __restrict__ o, size_t n) {
  size_t i = (size_t)blockIdx.x * blockDim.x + threadIdx.x;
  size_t s = (size_t)gridDim.x * blockDim.x;
  for (; i < n; i += s) o[i] = f2bf(a[i]);
}

// W [K, Nc] fp32 row-major -> Wt [Nc, K] bf16 (B columns contiguous in K)
__global__ void pack_wt(const float* __restrict__ Wsrc,
                        unsigned short* __restrict__ Wt, int K, int Nc) {
  size_t tot = (size_t)K * Nc;
  size_t i = (size_t)blockIdx.x * blockDim.x + threadIdx.x;
  size_t s = (size_t)gridDim.x * blockDim.x;
  for (; i < tot; i += s) {
    int k = (int)(i / Nc);
    int n = (int)(i % Nc);
    Wt[(size_t)n * K + k] = f2bf(Wsrc[i]);
  }
}

// ---------------- bf16 WMMA GEMM with async LDS double-buffering ----------------
// C[M,Nc] = A[M,K] * B[K,Nc].  A: bf16 row-major [Mpad,K]; Bt: bf16 [Nc,K].
// Block: 256 thr = 8 waves, tile 128(M) x 128(N). Wave (wm 0..3, wn 0..1): 32x64.
// Per K-step (32): stage A 128x32 + B 128x32 via GLOBAL_LOAD_ASYNC_TO_LDS_B128,
// double-buffered; 8 WMMAs per wave per step, all 12 fragment ds_loads issued
// up-front into distinct registers so DScnt waits are incremental.

union Frag { uint4 q[2]; v16bf v; };

#define LROW 40                      // padded LDS row stride (bf16 elems): 80B, conflict-free
#define LTILE (128 * LROW)           // elems per staged tile

__global__ void __launch_bounds__(256)
gemm_bf16_wmma(const unsigned short* __restrict__ A,
               const unsigned short* __restrict__ Bt,
               float* __restrict__ Cmat, int M, int K, int Nc) {
  __shared__ __align__(16) unsigned short ldsA[2][LTILE];
  __shared__ __align__(16) unsigned short ldsB[2][LTILE];

  const int Mblk = (M + 127) >> 7;
  const int bm = blockIdx.x % Mblk;
  const int bn = blockIdx.x / Mblk;
  const int tid  = threadIdx.x;
  const int lane = tid & 31;
  const int wave = tid >> 5;
  const int wm = wave & 3;           // 0..3 -> 32-row strip
  const int wn = wave >> 2;          // 0..1 -> 64-col strip
  const int half = lane >> 4;        // 0|1
  const int l16  = lane & 15;

  // per-thread async-copy assignments: 2 A chunks + 2 B chunks of 16B per K-step
  const unsigned short* aG[2];
  const unsigned short* bG[2];
  unsigned aOff[2], bOff[2];
  const unsigned baseA = (unsigned)(uintptr_t)&ldsA[0][0]; // low 32 bits = LDS offset
  const unsigned baseB = (unsigned)(uintptr_t)&ldsB[0][0];
#pragma unroll
  for (int i = 0; i < 2; ++i) {
    int chunk = tid + i * 256;       // 0..511
    int row = chunk >> 2;            // 0..127
    int q   = chunk & 3;             // 8-elem group within K-step
    aG[i] = A  + (size_t)(bm * 128 + row) * K + q * 8;
    bG[i] = Bt + (size_t)(bn * 128 + row) * K + q * 8;
    unsigned loff = (unsigned)(row * LROW + q * 8) * 2u;
    aOff[i] = baseA + loff;
    bOff[i] = baseB + loff;
  }
  const unsigned bufBytes = LTILE * 2u;

  auto issue = [&](int ks) {
    const int kk = ks << 5;
    const unsigned bsel = (unsigned)(ks & 1) * bufBytes;
#pragma unroll
    for (int i = 0; i < 2; ++i) {
      asm volatile("global_load_async_to_lds_b128 %0, %1, off"
                   :: "v"(aOff[i] + bsel), "v"(aG[i] + kk) : "memory");
      asm volatile("global_load_async_to_lds_b128 %0, %1, off"
                   :: "v"(bOff[i] + bsel), "v"(bG[i] + kk) : "memory");
    }
  };

  v8f acc[2][4] = {};
  const int nk = K >> 5;

  issue(0);
  for (int ks = 0; ks < nk; ++ks) {
    asm volatile("s_wait_asynccnt 0" ::: "memory"); // own async copies done
    __syncthreads();                               // => everyone's copies done;
                                                   // also: all waves finished reading
                                                   // the buffer we are about to refill
    if (ks + 1 < nk) issue(ks + 1);

    const int cur = ks & 1;
    // Hoist ALL fragment loads into distinct registers: 12 ds_load_b128 issued
    // together, WMMAs consume in load order -> incremental s_wait_dscnt.
    Frag fa[2];
    Frag fb[4];
#pragma unroll
    for (int mi = 0; mi < 2; ++mi) {
      // A 16x32 layout: elems 0..7 -> K = half*8+0..7 ; elems 8..15 -> K = 16+half*8+0..7
      const unsigned short* Ab = &ldsA[cur][(wm * 32 + mi * 16 + l16) * LROW + half * 8];
      fa[mi].q[0] = *(const uint4*)(Ab);
      fa[mi].q[1] = *(const uint4*)(Ab + 16);
    }
#pragma unroll
    for (int ni = 0; ni < 4; ++ni) {
      // B 32x16 layout: elems 0..7 -> K = half*8+0..7 ; elems 8..15 -> K = 16+half*8+0..7
      const unsigned short* Bb = &ldsB[cur][(wn * 64 + ni * 16 + l16) * LROW + half * 8];
      fb[ni].q[0] = *(const uint4*)(Bb);
      fb[ni].q[1] = *(const uint4*)(Bb + 16);
    }
#pragma unroll
    for (int ni = 0; ni < 4; ++ni)
#pragma unroll
      for (int mi = 0; mi < 2; ++mi)
        acc[mi][ni] = __builtin_amdgcn_wmma_f32_16x16x32_bf16(
            false, fa[mi].v, false, fb[ni].v, (short)0, acc[mi][ni], false, false);
  }

  // C/D f32 layout: elem r -> row += half*8 + r, col += l16
  const int rbase = bm * 128 + wm * 32 + half * 8;
  const int cbase = bn * 128 + wn * 64 + l16;
#pragma unroll
  for (int mi = 0; mi < 2; ++mi)
#pragma unroll
    for (int ni = 0; ni < 4; ++ni)
#pragma unroll
      for (int r = 0; r < 8; ++r) {
        int row = rbase + mi * 16 + r;
        if (row < M)
          Cmat[(size_t)row * Nc + (cbase + ni * 16)] = acc[mi][ni][r];
      }
}

// ---------------- attention ----------------

// one wave per (node, head): al_s[n,h] = <h[n,h,:], a_src[h,:]>, same for a_dst
__global__ void attn_logits(const float* __restrict__ h,
                            const float* __restrict__ a_src,
                            const float* __restrict__ a_dst,
                            float* __restrict__ al_s, float* __restrict__ al_d,
                            int N, int C) {
  int n = blockIdx.x;
  int hd = threadIdx.x >> 5;
  int lane = threadIdx.x & 31;
  const float* hp = h + (size_t)n * HEADS * C + (size_t)hd * C;
  float ss = 0.0f, sd = 0.0f;
  for (int c = lane; c < C; c += 32) {
    float v = hp[c];
    ss += v * a_src[hd * C + c];
    sd += v * a_dst[hd * C + c];
  }
#pragma unroll
  for (int o = 16; o > 0; o >>= 1) {
    ss += __shfl_down(ss, o, 32);
    sd += __shfl_down(sd, o, 32);
  }
  if (lane == 0) {
    al_s[n * HEADS + hd] = ss;
    al_d[n * HEADS + hd] = sd;
  }
}

// per (edge, head): e = leaky_relu(al_s[src]+al_d[dst]); store; atomic max into m[dst]
__global__ void edge_logits_max(const int* __restrict__ srcE, const int* __restrict__ dstE,
                                int E, int N,
                                const float* __restrict__ al_s, const float* __restrict__ al_d,
                                float* __restrict__ eb, float* __restrict__ m) {
  int Etot = E + N;
  int idx = blockIdx.x * blockDim.x + threadIdx.x;
  if (idx >= Etot * HEADS) return;
  int e = idx / HEADS, hd = idx % HEADS;
  int s = (e < E) ? srcE[e] : (e - E);      // appended self-loops
  int d = (e < E) ? dstE[e] : (e - E);
  float v = al_s[s * HEADS + hd] + al_d[d * HEADS + hd];
  v = (v > 0.0f) ? v : 0.2f * v;            // leaky_relu, slope 0.2
  eb[idx] = v;
  atomicMaxF32(&m[d * HEADS + hd], v);
}

// ex = exp(e - m[dst]); eb <- ex; denom[dst] += ex
__global__ void edge_exp_sum(const int* __restrict__ dstE, int E, int N,
                             const float* __restrict__ m,
                             float* __restrict__ eb, float* __restrict__ denom) {
  int Etot = E + N;
  int idx = blockIdx.x * blockDim.x + threadIdx.x;
  if (idx >= Etot * HEADS) return;
  int e = idx / HEADS, hd = idx % HEADS;
  int d = (e < E) ? dstE[e] : (e - E);
  float ex = expf(eb[idx] - m[d * HEADS + hd]);
  eb[idx] = ex;
  atomicAdd(&denom[d * HEADS + hd], ex);
}

// acc[dst, h*C+c] += alpha[e,h] * hbuf[src, h*C+c]; one block per edge
__global__ void edge_aggregate(const int* __restrict__ srcE, const int* __restrict__ dstE,
                               int E, int N,
                               const float* __restrict__ eb, const float* __restrict__ denom,
                               const float* __restrict__ hbuf, float* __restrict__ acc,
                               int C) {
  int e = blockIdx.x;
  int s = (e < E) ? srcE[e] : (e - E);
  int d = (e < E) ? dstE[e] : (e - E);
  __shared__ float salpha[HEADS];
  if (threadIdx.x < HEADS)
    salpha[threadIdx.x] =
        eb[(size_t)e * HEADS + threadIdx.x] /
        (denom[d * HEADS + threadIdx.x] + 1e-16f);
  __syncthreads();
  int HC = HEADS * C;
  const float* hs = hbuf + (size_t)s * HC;
  float* ad = acc + (size_t)d * HC;
  for (int c = threadIdx.x; c < HC; c += blockDim.x)
    atomicAdd(&ad[c], salpha[c / C] * hs[c]);
}

// ---------------- epilogues ----------------

__global__ void finalize_concat(float* __restrict__ acc, const float* __restrict__ bias,
                                size_t tot, int HC) {
  size_t i = (size_t)blockIdx.x * blockDim.x + threadIdx.x;
  size_t s = (size_t)gridDim.x * blockDim.x;
  for (; i < tot; i += s) acc[i] = eluf(acc[i] + bias[i % HC]);
}

// layer 5: mean over heads + bias + elu, then scatter into per-graph pool
__global__ void finalize_mean_pool(const float* __restrict__ acc,
                                   const float* __restrict__ bias,
                                   const int* __restrict__ batch,
                                   float* __restrict__ pool, float* __restrict__ cnt,
                                   int N, int C) {
  int n = blockIdx.x;
  int g = batch[n];
  for (int c = threadIdx.x; c < C; c += blockDim.x) {
    float sum = 0.0f;
#pragma unroll
    for (int hd = 0; hd < HEADS; ++hd)
      sum += acc[(size_t)n * HEADS * C + (size_t)hd * C + c];
    float v = eluf(sum * (1.0f / HEADS) + bias[c]);
    atomicAdd(&pool[g * C + c], v);
  }
  if (threadIdx.x == 0) atomicAdd(&cnt[g], 1.0f);
}

// pooled = elu(pool/max(cnt,1)); out = pooled @ W_lin + b_lin   (64 x 256 x 128)
__global__ void pool_linear(const float* __restrict__ pool, const float* __restrict__ cnt,
                            const float* __restrict__ Wl, const float* __restrict__ bl,
                            float* __restrict__ out) {
  int g = blockIdx.x;     // 0..63
  int o = threadIdx.x;    // 0..127
  float c = cnt[g];
  c = (c < 1.0f) ? 1.0f : c;
  float inv = 1.0f / c;
  float s = 0.0f;
  for (int k = 0; k < 256; ++k) {
    float p = eluf(pool[g * 256 + k] * inv);
    s += p * Wl[k * 128 + o];
  }
  out[g * 128 + o] = s + bl[o];
}

// ---------------- host orchestration ----------------

static inline unsigned gsblocks(size_t n) {
  size_t b = (n + 255) / 256;
  return (unsigned)(b > 8192 ? 8192 : b);
}

extern "C" void kernel_launch(void* const* d_in, const int* in_sizes, int n_in,
                              void* d_out, int out_size, void* d_ws, size_t ws_size,
                              hipStream_t stream) {
  const float* x     = (const float*)d_in[0];
  const int*   eidx  = (const int*)d_in[1];
  const int*   batch = (const int*)d_in[2];
  const int N = in_sizes[0] / 128;
  const int E = in_sizes[1] / 2;
  const int* srcE = eidx;
  const int* dstE = eidx + E;

  const float *Wm[5], *as[5], *ad[5], *bs[5];
  for (int i = 0; i < 5; ++i) {
    Wm[i] = (const float*)d_in[3 + 4 * i];
    as[i] = (const float*)d_in[4 + 4 * i];
    ad[i] = (const float*)d_in[5 + 4 * i];
    bs[i] = (const float*)d_in[6 + 4 * i];
  }
  const float* Wl = (const float*)d_in[23];
  const float* bl = (const float*)d_in[24];

  // workspace carve-up
  char* w = (char*)d_ws;
  auto carve = [&](size_t bytes) -> void* {
    void* p = (void*)w;
    w += (bytes + 255) & ~(size_t)255;
    return p;
  };
  const size_t MAXF = (size_t)N * 2048;
  const int Mpad = ((N + 127) / 128) * 128;         // A staging rows padded for 128-tiles
  float* big0  = (float*)carve(MAXF * 4);           // rotate: input / acc
  float* big1  = (float*)carve(MAXF * 4);
  float* hbuf  = (float*)carve(MAXF * 4);           // h = x @ W
  unsigned short* abf  = (unsigned short*)carve((size_t)Mpad * 2048 * 2);   // bf16 A
  unsigned short* wtbf = (unsigned short*)carve((size_t)2048 * 2048 * 2);   // bf16 W^T
  float* al_s  = (float*)carve((size_t)N * HEADS * 4);
  float* al_d  = (float*)carve((size_t)N * HEADS * 4);
  float* mbuf  = (float*)carve((size_t)N * HEADS * 4);
  float* denom = (float*)carve((size_t)N * HEADS * 4);
  float* eb    = (float*)carve((size_t)(E + N) * HEADS * 4);
  float* pool  = (float*)carve((size_t)64 * 256 * 4);
  float* cnt   = (float*)carve((size_t)64 * 4);

  const int Kd[5] = {128, 512, 2048, 2048, 2048};
  const int Cd[5] = {64, 256, 256, 256, 256};
  const int Etot = E + N;
  const int eThreads = Etot * HEADS;

  const float* inF = x;
  for (int L = 0; L < 5; ++L) {
    const int K = Kd[L], C = Cd[L], HC = HEADS * C;
    float* acc = (L % 2 == 0) ? big0 : big1;

    // pack operands to bf16
    pack_bf16<<<gsblocks((size_t)N * K), 256, 0, stream>>>(inF, abf, (size_t)N * K);
    pack_wt<<<gsblocks((size_t)K * HC), 256, 0, stream>>>(Wm[L], wtbf, K, HC);

    // GEMM: h = A * W  (N x K x HC); 128x128 block tiles, async double-buffered LDS
    {
      int Mblk = (N + 127) / 128;
      int Nblk = HC / 128;
      gemm_bf16_wmma<<<Mblk * Nblk, 256, 0, stream>>>(abf, wtbf, hbuf, N, K, HC);
    }

    // attention logits per (node, head)
    attn_logits<<<N, HEADS * 32, 0, stream>>>(hbuf, as[L], ad[L], al_s, al_d, N, C);

    // segment softmax state
    fill_f32<<<gsblocks((size_t)N * HEADS), 256, 0, stream>>>(mbuf, -INFINITY, (size_t)N * HEADS);
    fill_f32<<<gsblocks((size_t)N * HEADS), 256, 0, stream>>>(denom, 0.0f, (size_t)N * HEADS);
    fill_f32<<<gsblocks((size_t)N * HC), 256, 0, stream>>>(acc, 0.0f, (size_t)N * HC);

    edge_logits_max<<<(eThreads + 255) / 256, 256, 0, stream>>>(srcE, dstE, E, N, al_s, al_d, eb, mbuf);
    edge_exp_sum<<<(eThreads + 255) / 256, 256, 0, stream>>>(dstE, E, N, mbuf, eb, denom);
    edge_aggregate<<<Etot, 256, 0, stream>>>(srcE, dstE, E, N, eb, denom, hbuf, acc, C);

    if (L < 4) {
      finalize_concat<<<gsblocks((size_t)N * HC), 256, 0, stream>>>(acc, bs[L], (size_t)N * HC, HC);
      inF = acc;
    } else {
      fill_f32<<<gsblocks(64 * 256), 256, 0, stream>>>(pool, 0.0f, (size_t)64 * 256);
      fill_f32<<<1, 64, 0, stream>>>(cnt, 0.0f, (size_t)64);
      finalize_mean_pool<<<N, 256, 0, stream>>>(acc, bs[L], batch, pool, cnt, N, C);
    }
  }

  pool_linear<<<64, 128, 0, stream>>>(pool, cnt, Wl, bl, (float*)d_out);
}
